// Qwen2MoeDecoderLayer_16587163697447
// MI455X (gfx1250) — compile-verified, matrix operand order checked by reading, not run on hardware
//
#include <hip/hip_runtime.h>
#include <cstddef>

// ---------------- dims ----------------
#define DH   2048     // hidden size
#define SQ   1024     // sequence length
#define NH   16       // attention heads
#define NKV  4        // kv heads
#define HD   128      // head dim
#define NE   8        // experts
#define IM   1408     // moe intermediate
#define SIM  5632     // shared intermediate

typedef __attribute__((ext_vector_type(16))) _Float16 v16h;
typedef __attribute__((ext_vector_type(8)))  _Float16 v8h;
typedef __attribute__((ext_vector_type(8)))  float    v8f;
typedef __attribute__((ext_vector_type(4)))  float    f32x4;

// ---------------------------------------------------------------------------
// Per-lane 16x32 f16 tile load (A-layout == B-layout).  CDNA5 WMMA 16-bit
// layout:
//   lanes 0-15  : row = lane,    K = 0..7 / 16..23
//   lanes 16-31 : row = lane-16, K = 8..15 / 24..31
// => per lane: two contiguous runs of 8 elements at k0+8g and k0+8g+16.
// f16 source: two 16 B loads.  f32 source: four 16 B loads + cvt.
// ---------------------------------------------------------------------------
__device__ __forceinline__ v16h load_tile16x32(const _Float16* __restrict__ p,
                                               size_t ld, int row0, int k0,
                                               int lane) {
  int g = (lane >> 4) & 1;
  const _Float16* r =
      p + (size_t)(row0 + (lane & 15)) * ld + (size_t)(k0 + 8 * g);
  v8h lo = *(const v8h*)(r);
  v8h hi = *(const v8h*)(r + 16);
  return __builtin_shufflevector(lo, hi, 0, 1, 2, 3, 4, 5, 6, 7,
                                 8, 9, 10, 11, 12, 13, 14, 15);
}

// B tile fetch from an LDS f16 panel stored as [rows][32] (64 B row stride).
__device__ __forceinline__ v16h lds_tile(const _Float16* __restrict__ panel,
                                         int row0, int g, int l15) {
  const _Float16* p = panel + (size_t)(row0 + l15) * 32 + 8 * g;
  v8h lo = *(const v8h*)(p);
  v8h hi = *(const v8h*)(p + 16);
  return __builtin_shufflevector(lo, hi, 0, 1, 2, 3, 4, 5, 6, 7,
                                 8, 9, 10, 11, 12, 13, 14, 15);
}

// Stage 16 consecutive fp32 weights -> 16 f16 into LDS (two 16 B stores).
__device__ __forceinline__ void stage16(const float* __restrict__ src,
                                        _Float16* __restrict__ dst) {
  f32x4 a = *(const f32x4*)(src);
  f32x4 b = *(const f32x4*)(src + 4);
  f32x4 c = *(const f32x4*)(src + 8);
  f32x4 d = *(const f32x4*)(src + 12);
  v8h lo, hi;
  lo[0] = (_Float16)a[0]; lo[1] = (_Float16)a[1];
  lo[2] = (_Float16)a[2]; lo[3] = (_Float16)a[3];
  lo[4] = (_Float16)b[0]; lo[5] = (_Float16)b[1];
  lo[6] = (_Float16)b[2]; lo[7] = (_Float16)b[3];
  hi[0] = (_Float16)c[0]; hi[1] = (_Float16)c[1];
  hi[2] = (_Float16)c[2]; hi[3] = (_Float16)c[3];
  hi[4] = (_Float16)d[0]; hi[5] = (_Float16)d[1];
  hi[6] = (_Float16)d[2]; hi[7] = (_Float16)d[3];
  *(v8h*)(dst) = lo;
  *(v8h*)(dst + 8) = hi;
}

__device__ __forceinline__ v8f wmma16(v16h a, v16h b, v8f c) {
  return __builtin_amdgcn_wmma_f32_16x16x32_f16(false, a, false, b,
                                                (short)0, c, false, false);
}

// ---------------------------------------------------------------------------
// GEMM: C[M,N] = op( A[M,K](f16) @ W[N,K](f32)^T ).  256 threads = 8 waves
// arranged 4 (M) x 2 (N); block tile 128x128; wave tile 32x64 (8 WMMA/k-step).
// B panel (128x32) staged to LDS as f16 once per block per k-step.
// Epilogue: +bias[n], *rowscale[m*rs_stride], +residual[m,n], optional +=C.
// Requires M % 128 == 0, N % 128 == 0, K % 32 == 0.
// ---------------------------------------------------------------------------
template <typename TC>
__global__ __launch_bounds__(256) void gemm_wmma_kernel(
    const _Float16* __restrict__ A, const float* __restrict__ W,
    TC* __restrict__ C, int M, int N, int K,
    const float* __restrict__ bias, const float* __restrict__ residual,
    const float* __restrict__ rowscale, int rs_stride, int accumulate) {
  __shared__ _Float16 Bs[128 * 32];
  int tid = threadIdx.x;
  int wave = tid >> 5, lane = tid & 31;
  int mw = wave & 3, nw = wave >> 2;
  int m0 = blockIdx.x * 128 + mw * 32;
  int n0b = blockIdx.y * 128;
  int g = lane >> 4, l15 = lane & 15;

  int srow = tid >> 1, shalf = tid & 1;
  const float* wsrc = W + (size_t)(n0b + srow) * K + 16 * shalf;
  _Float16* sdst = &Bs[srow * 32 + 16 * shalf];

  v8f acc[2][4] = {};
  for (int k0 = 0; k0 < K; k0 += 32) {
    stage16(wsrc + k0, sdst);
    __builtin_prefetch(wsrc + k0 + 32, 0, 0);
    __syncthreads();
    v16h a0 = load_tile16x32(A, (size_t)K, m0, k0, lane);
    v16h a1 = load_tile16x32(A, (size_t)K, m0 + 16, k0, lane);
    for (int t = 0; t < 4; ++t) {
      v16h b = lds_tile(Bs, nw * 64 + t * 16, g, l15);
      acc[0][t] = wmma16(a0, b, acc[0][t]);
      acc[1][t] = wmma16(a1, b, acc[1][t]);
    }
    __syncthreads();
  }

  for (int t = 0; t < 4; ++t) {
    int col = n0b + nw * 64 + 16 * t + l15;
    float bv = bias ? bias[col] : 0.f;
    for (int sub = 0; sub < 2; ++sub)
      for (int r = 0; r < 8; ++r) {
        int row = m0 + sub * 16 + r + 8 * g;
        float v = acc[sub][t][r] + bv;
        if (rowscale) v *= rowscale[(size_t)row * rs_stride];
        if (residual) v += residual[(size_t)row * N + col];
        TC* cp = &C[(size_t)row * N + col];
        *cp = accumulate ? (TC)((float)*cp + v) : (TC)v;
      }
  }
}

// ---------------------------------------------------------------------------
// Fused gate/up + SiLU: C[M,N](f16) = silu(A@Wg^T) * (A@Wu^T).
// Same tiling; two LDS panels, two accumulator sets.
// ---------------------------------------------------------------------------
__global__ __launch_bounds__(256) void gateup_silu_kernel(
    const _Float16* __restrict__ A, const float* __restrict__ Wg,
    const float* __restrict__ Wu, _Float16* __restrict__ C,
    int M, int N, int K) {
  __shared__ _Float16 Bg[128 * 32];
  __shared__ _Float16 Bu[128 * 32];
  int tid = threadIdx.x;
  int wave = tid >> 5, lane = tid & 31;
  int mw = wave & 3, nw = wave >> 2;
  int m0 = blockIdx.x * 128 + mw * 32;
  int n0b = blockIdx.y * 128;
  int g = lane >> 4, l15 = lane & 15;

  int srow = tid >> 1, shalf = tid & 1;
  const float* gsrc = Wg + (size_t)(n0b + srow) * K + 16 * shalf;
  const float* usrc = Wu + (size_t)(n0b + srow) * K + 16 * shalf;
  _Float16* gdst = &Bg[srow * 32 + 16 * shalf];
  _Float16* udst = &Bu[srow * 32 + 16 * shalf];

  v8f accg[2][4] = {};
  v8f accu[2][4] = {};
  for (int k0 = 0; k0 < K; k0 += 32) {
    stage16(gsrc + k0, gdst);
    stage16(usrc + k0, udst);
    __builtin_prefetch(gsrc + k0 + 32, 0, 0);
    __builtin_prefetch(usrc + k0 + 32, 0, 0);
    __syncthreads();
    v16h a0 = load_tile16x32(A, (size_t)K, m0, k0, lane);
    v16h a1 = load_tile16x32(A, (size_t)K, m0 + 16, k0, lane);
    for (int t = 0; t < 4; ++t) {
      v16h bg = lds_tile(Bg, nw * 64 + t * 16, g, l15);
      v16h bu = lds_tile(Bu, nw * 64 + t * 16, g, l15);
      accg[0][t] = wmma16(a0, bg, accg[0][t]);
      accg[1][t] = wmma16(a1, bg, accg[1][t]);
      accu[0][t] = wmma16(a0, bu, accu[0][t]);
      accu[1][t] = wmma16(a1, bu, accu[1][t]);
    }
    __syncthreads();
  }

  for (int t = 0; t < 4; ++t) {
    int col = n0b + nw * 64 + 16 * t + l15;
    for (int sub = 0; sub < 2; ++sub)
      for (int r = 0; r < 8; ++r) {
        int row = m0 + sub * 16 + r + 8 * g;
        float gv = accg[sub][t][r];
        float uv = accu[sub][t][r];
        C[(size_t)row * N + col] =
            (_Float16)(gv / (1.0f + __expf(-gv)) * uv);
      }
  }
}

// ---------------------------------------------------------------------------
// RMSNorm per row: fp32 in, f16 out (activation path is f16 end-to-end).
// ---------------------------------------------------------------------------
__global__ __launch_bounds__(256) void rmsnorm_kernel(
    const float* __restrict__ x, const float* __restrict__ w,
    _Float16* __restrict__ y) {
  __shared__ float red[256];
  int row = blockIdx.x;
  const float* xr = x + (size_t)row * DH;
  float s = 0.f;
  for (int i = threadIdx.x; i < DH; i += 256) { float v = xr[i]; s += v * v; }
  red[threadIdx.x] = s;
  __syncthreads();
  for (int off = 128; off > 0; off >>= 1) {
    if (threadIdx.x < off) red[threadIdx.x] += red[threadIdx.x + off];
    __syncthreads();
  }
  float inv = rsqrtf(red[0] / (float)DH + 1e-6f);
  for (int i = threadIdx.x; i < DH; i += 256)
    y[(size_t)row * DH + i] = (_Float16)(w[i] * xr[i] * inv);
}

// ---------------------------------------------------------------------------
// RoPE in-place on f16 [SQ, nheads*HD], theta = 1e6
// ---------------------------------------------------------------------------
__global__ void rope_kernel(_Float16* __restrict__ x, int nheads) {
  int idx = blockIdx.x * blockDim.x + threadIdx.x;
  int total = SQ * nheads * 64;
  if (idx >= total) return;
  int i = idx & 63;
  int t = idx >> 6;
  int hh = t % nheads;
  int s  = t / nheads;
  float invf = __powf(1.0e6f, -(float)i / 64.0f);
  float ang = (float)s * invf;
  float c = __cosf(ang), sn = __sinf(ang);
  size_t base = (size_t)s * nheads * HD + (size_t)hh * HD + i;
  float x0 = (float)x[base], x1 = (float)x[base + 64];
  x[base]      = (_Float16)(x0 * c - x1 * sn);
  x[base + 64] = (_Float16)(x1 * c + x0 * sn);
}

// v [SQ, NKV*HD] -> vt [NKV*HD, SQ]  (f16)
__global__ void transpose_kernel(const _Float16* __restrict__ v,
                                 _Float16* __restrict__ vt) {
  int idx = blockIdx.x * blockDim.x + threadIdx.x;
  if (idx >= SQ * NKV * HD) return;
  int r = idx / (NKV * HD), c = idx % (NKV * HD);
  vt[(size_t)c * SQ + r] = v[idx];
}

// ---------------------------------------------------------------------------
// Flash attention: one wave per (q-tile, head). Online softmax, P transposed
// D-layout -> A-layout through a 1 KB LDS tile, P@V via WMMA against Vt.
// q/k/vt are f16; output f16.
// ---------------------------------------------------------------------------
__global__ __launch_bounds__(32) void attn_kernel(
    const _Float16* __restrict__ q, const _Float16* __restrict__ k,
    const _Float16* __restrict__ vt, _Float16* __restrict__ out) {
  __shared__ _Float16 Sp[16][32];
  int lane = threadIdx.x & 31;
  int h  = blockIdx.y;
  int q0 = blockIdx.x * 16;
  int kvh = h >> 2;
  int g = lane >> 4, l15 = lane & 15;

  v16h qa[4];
  for (int c = 0; c < 4; ++c)
    qa[c] = load_tile16x32(q + (size_t)h * HD, DH, q0, c * 32, lane);

  v8f acc[8] = {};
  float mx[8], lsum[8];
  for (int r = 0; r < 8; ++r) { mx[r] = -3.0e38f; lsum[r] = 0.f; }
  const float scale = 0.08838834764831845f;  // 1/sqrt(128)

  for (int kc = 0; kc <= q0 + 15; kc += 32) {
    v8f s[2] = {};
    for (int sub = 0; sub < 2; ++sub)
      for (int c = 0; c < 4; ++c) {
        v16h kb = load_tile16x32(k + (size_t)kvh * HD, (size_t)NKV * HD,
                                 kc + sub * 16, c * 32, lane);
        s[sub] = wmma16(qa[c], kb, s[sub]);
      }
    for (int sub = 0; sub < 2; ++sub) {
      int col = kc + sub * 16 + l15;
      for (int r = 0; r < 8; ++r) {
        int row = q0 + r + 8 * g;
        float v = s[sub][r] * scale;
        s[sub][r] = (col <= row) ? v : -3.0e38f;
      }
    }
    float pr0[8], pr1[8];
    for (int r = 0; r < 8; ++r) {
      float v = fmaxf(s[0][r], s[1][r]);
      v = fmaxf(v, __shfl_xor(v, 1));
      v = fmaxf(v, __shfl_xor(v, 2));
      v = fmaxf(v, __shfl_xor(v, 4));
      v = fmaxf(v, __shfl_xor(v, 8));
      float nm = fmaxf(mx[r], v);
      float fs = __expf(mx[r] - nm);
      mx[r] = nm;
      float p0 = __expf(s[0][r] - nm);
      float p1 = __expf(s[1][r] - nm);
      pr0[r] = p0; pr1[r] = p1;
      float rs = p0 + p1;
      rs += __shfl_xor(rs, 1);
      rs += __shfl_xor(rs, 2);
      rs += __shfl_xor(rs, 4);
      rs += __shfl_xor(rs, 8);
      lsum[r] = lsum[r] * fs + rs;
      for (int t = 0; t < 8; ++t) acc[t][r] *= fs;
    }
    __syncthreads();
    for (int r = 0; r < 8; ++r) {
      Sp[r + 8 * g][l15]      = (_Float16)pr0[r];
      Sp[r + 8 * g][16 + l15] = (_Float16)pr1[r];
    }
    __syncthreads();
    v16h pa;
    {
      const _Float16* prow = &Sp[l15][0];
      for (int j = 0; j < 8; ++j) pa[j]     = prow[8 * g + j];
      for (int j = 0; j < 8; ++j) pa[8 + j] = prow[8 * g + 16 + j];
    }
    for (int t = 0; t < 8; ++t) {
      v16h vb = load_tile16x32(vt + (size_t)kvh * HD * SQ, SQ, t * 16, kc,
                               lane);
      acc[t] = wmma16(pa, vb, acc[t]);
    }
    __syncthreads();
  }
  for (int r = 0; r < 8; ++r) {
    float inv = 1.0f / lsum[r];
    int row = q0 + r + 8 * g;
    for (int t = 0; t < 8; ++t)
      out[(size_t)row * DH + (size_t)h * HD + t * 16 + l15] =
          (_Float16)(acc[t][r] * inv);
  }
}

// ---------------------------------------------------------------------------
// Router: logits -> softmax -> top-2 -> dense weights [SQ, NE]
// ---------------------------------------------------------------------------
__global__ __launch_bounds__(256) void router_kernel(
    const _Float16* __restrict__ x2, const float* __restrict__ rw,
    float* __restrict__ dw) {
  __shared__ float part[256];
  int t = blockIdx.x;
  int e  = threadIdx.x & 7;
  int ch = threadIdx.x >> 3;
  const _Float16* xr = x2 + (size_t)t * DH;
  const float* wr = rw + (size_t)e * DH;
  float s = 0.f;
  for (int i = ch * 64; i < ch * 64 + 64; ++i) s += (float)xr[i] * wr[i];
  part[threadIdx.x] = s;
  __syncthreads();
  for (int off = 128; off >= 8; off >>= 1) {
    if (threadIdx.x < off) part[threadIdx.x] += part[threadIdx.x + off];
    __syncthreads();
  }
  if (threadIdx.x == 0) {
    float lg[NE], mxv = -3.0e38f;
    for (int i = 0; i < NE; ++i) { lg[i] = part[i]; mxv = fmaxf(mxv, lg[i]); }
    float sum = 0.f;
    for (int i = 0; i < NE; ++i) { lg[i] = __expf(lg[i] - mxv); sum += lg[i]; }
    int i1 = 0;
    for (int i = 1; i < NE; ++i) if (lg[i] > lg[i1]) i1 = i;
    int i2 = (i1 == 0) ? 1 : 0;
    for (int i = 0; i < NE; ++i) if (i != i1 && lg[i] > lg[i2]) i2 = i;
    for (int i = 0; i < NE; ++i) {
      float p = (i == i1 || i == i2) ? (lg[i] / sum) : 0.f;
      dw[(size_t)t * NE + i] = p;
    }
  }
}

__global__ __launch_bounds__(256) void sgate_kernel(
    const _Float16* __restrict__ x2, const float* __restrict__ w,
    float* __restrict__ sg) {
  __shared__ float red[256];
  int t = blockIdx.x;
  const _Float16* xr = x2 + (size_t)t * DH;
  float s = 0.f;
  for (int i = threadIdx.x; i < DH; i += 256) s += (float)xr[i] * w[i];
  red[threadIdx.x] = s;
  __syncthreads();
  for (int off = 128; off > 0; off >>= 1) {
    if (threadIdx.x < off) red[threadIdx.x] += red[threadIdx.x + off];
    __syncthreads();
  }
  if (threadIdx.x == 0) sg[t] = 1.0f / (1.0f + __expf(-red[0]));
}

__global__ void final_kernel(const float* __restrict__ res2,
                             const float* __restrict__ moe,
                             const float* __restrict__ sh,
                             const float* __restrict__ sg,
                             float* __restrict__ out) {
  int i = blockIdx.x * blockDim.x + threadIdx.x;
  if (i >= SQ * DH) return;
  int t = i >> 11;  // / DH
  out[i] = res2[i] + moe[i] + sg[t] * sh[i];
}

// ---------------------------------------------------------------------------
extern "C" void kernel_launch(void* const* d_in, const int* in_sizes, int n_in,
                              void* d_out, int out_size, void* d_ws,
                              size_t ws_size, hipStream_t stream) {
  (void)in_sizes; (void)n_in; (void)out_size; (void)ws_size;
  const float* hidden = (const float*)d_in[0];
  const float* q_w  = (const float*)d_in[2];
  const float* q_b  = (const float*)d_in[3];
  const float* k_w  = (const float*)d_in[4];
  const float* k_b  = (const float*)d_in[5];
  const float* v_w  = (const float*)d_in[6];
  const float* v_b  = (const float*)d_in[7];
  const float* o_w  = (const float*)d_in[8];
  const float* ln1  = (const float*)d_in[9];
  const float* ln2  = (const float*)d_in[10];
  const float* rt_w = (const float*)d_in[11];
  const float* eg   = (const float*)d_in[12];
  const float* eu   = (const float*)d_in[13];
  const float* ed   = (const float*)d_in[14];
  const float* sgw  = (const float*)d_in[15];
  const float* suw  = (const float*)d_in[16];
  const float* sdw  = (const float*)d_in[17];
  const float* segw = (const float*)d_in[18];
  float* out = (float*)d_out;

  size_t off = 0;
  char* base = (char*)d_ws;
  auto alloc = [&](size_t bytes) {
    char* p = base + off;
    off += (bytes + 63) & ~(size_t)63;
    return p;
  };
  _Float16* xn   = (_Float16*)alloc((size_t)SQ * DH * 2);
  _Float16* qb   = (_Float16*)alloc((size_t)SQ * DH * 2);
  _Float16* kb   = (_Float16*)alloc((size_t)SQ * NKV * HD * 2);
  _Float16* vb   = (_Float16*)alloc((size_t)SQ * NKV * HD * 2);
  _Float16* vtb  = (_Float16*)alloc((size_t)SQ * NKV * HD * 2);
  _Float16* ao   = (_Float16*)alloc((size_t)SQ * DH * 2);
  _Float16* x2   = (_Float16*)alloc((size_t)SQ * DH * 2);
  _Float16* hbuf = (_Float16*)alloc((size_t)SQ * SIM * 2);
  float* hmid = (float*)alloc((size_t)SQ * DH * 4);
  float* dw   = (float*)alloc((size_t)SQ * NE * 4);
  float* sg   = (float*)alloc((size_t)SQ * 4);
  float* moe  = (float*)alloc((size_t)SQ * DH * 4);
  float* shb  = (float*)alloc((size_t)SQ * DH * 4);

  auto gemm_h = [&](const _Float16* A, const float* W, _Float16* C, int M,
                    int N, int K, const float* bias) {
    dim3 grid(M / 128, N / 128);
    gemm_wmma_kernel<_Float16><<<grid, 256, 0, stream>>>(
        A, W, C, M, N, K, bias, nullptr, nullptr, 0, 0);
  };
  auto gemm_f = [&](const _Float16* A, const float* W, float* C, int M, int N,
                    int K, const float* residual, const float* rowscale,
                    int rs_stride, int accum) {
    dim3 grid(M / 128, N / 128);
    gemm_wmma_kernel<float><<<grid, 256, 0, stream>>>(
        A, W, C, M, N, K, nullptr, residual, rowscale, rs_stride, accum);
  };
  auto gateup = [&](const _Float16* A, const float* Wg, const float* Wu,
                    _Float16* C, int M, int N, int K) {
    dim3 grid(M / 128, N / 128);
    gateup_silu_kernel<<<grid, 256, 0, stream>>>(A, Wg, Wu, C, M, N, K);
  };

  // ---- attention block ----
  rmsnorm_kernel<<<SQ, 256, 0, stream>>>(hidden, ln1, xn);
  gemm_h(xn, q_w, qb, SQ, DH, DH, q_b);
  gemm_h(xn, k_w, kb, SQ, NKV * HD, DH, k_b);
  gemm_h(xn, v_w, vb, SQ, NKV * HD, DH, v_b);
  rope_kernel<<<(SQ * NH * 64 + 255) / 256, 256, 0, stream>>>(qb, NH);
  rope_kernel<<<(SQ * NKV * 64 + 255) / 256, 256, 0, stream>>>(kb, NKV);
  transpose_kernel<<<(SQ * NKV * HD + 255) / 256, 256, 0, stream>>>(vb, vtb);
  attn_kernel<<<dim3(SQ / 16, NH), 32, 0, stream>>>(qb, kb, vtb, ao);
  gemm_f(ao, o_w, hmid, SQ, DH, DH, hidden, nullptr, 0, 0);

  // ---- MoE block ----
  rmsnorm_kernel<<<SQ, 256, 0, stream>>>(hmid, ln2, x2);
  router_kernel<<<SQ, 256, 0, stream>>>(x2, rt_w, dw);

  for (int e = 0; e < NE; ++e) {
    const float* ge = eg + (size_t)e * IM * DH;
    const float* ue = eu + (size_t)e * IM * DH;
    const float* de = ed + (size_t)e * DH * IM;
    gateup(x2, ge, ue, hbuf, SQ, IM, DH);
    // moe (+)= dense_w[:,e] * (hbuf @ down_e^T)
    gemm_f(hbuf, de, moe, SQ, DH, IM, nullptr, dw + e, NE, e > 0);
  }

  // shared expert
  gateup(x2, sgw, suw, hbuf, SQ, SIM, DH);
  gemm_f(hbuf, sdw, shb, SQ, DH, SIM, nullptr, nullptr, 0, 0);
  sgate_kernel<<<SQ, 256, 0, stream>>>(x2, segw, sg);

  final_kernel<<<(SQ * DH + 255) / 256, 256, 0, stream>>>(hmid, moe, shb, sg,
                                                          out);
}
